// Mask_20289425506515
// MI455X (gfx1250) — matrix operand order repaired and checked by reading, not compiled, not run on hardware
//
#include <hip/hip_runtime.h>
#include <hip/hip_bf16.h>

typedef __attribute__((ext_vector_type(16))) __bf16        v16bf;
typedef __attribute__((ext_vector_type(8)))  float         v8f;
typedef __attribute__((ext_vector_type(8)))  unsigned int  v8u;

#define TM 128
#define TK 32
#define TN 32
#define ASTRIDE 20   // dwords per A row (16 packed + 4 pad) -> 16B aligned rows, no bank conflicts
#define BSTRIDE 20   // dwords per B col

static __device__ __forceinline__ unsigned int pk_bf16(float lo, float hi) {
    union { __bf16 h[2]; unsigned int u; } r;
    r.h[0] = (__bf16)lo;
    r.h[1] = (__bf16)hi;
    return r.u;
}

// ---------------------------------------------------------------------------
// Tiled bf16 WMMA GEMM:  C[N x M] = A[N x K] * W[K x M] (+ bias)
// Block = 256 threads (8 waves). Block tile = 128 rows x 32 cols.
// LDS tiles are stored K-pair-packed (2 bf16 / dword) in fragment order so
// each fragment is exactly two 16B-aligned uint4 LDS reads (ds_load_b128).
// ---------------------------------------------------------------------------
__global__ __launch_bounds__(256) void gemm_bf16_wmma(
    const float* __restrict__ A, const float* __restrict__ W,
    const float* __restrict__ bias, float* __restrict__ C,
    int Nrows, int K, int M)
{
    __shared__ unsigned int AsU[TM * ASTRIDE];   // AsU[row][k2], k2 = K/2 within tile
    __shared__ unsigned int BsU[TN * BSTRIDE];   // BsU[col][k2]  (column-major packed!)

    const int tid  = threadIdx.x;
    const int wave = tid >> 5;
    const int lane = tid & 31;
    const int row0 = blockIdx.x * TM;
    const int col0 = blockIdx.y * TN;
    const bool interior = (row0 + TM) <= Nrows;

    const int half = lane >> 4;               // K sub-range selector
    const int arow = (wave << 4) + (lane & 15);
    const int bcol = lane & 15;

    v8f acc0 = {};
    v8f acc1 = {};

    for (int kt = 0; kt < K; kt += TK) {
        // ---- stage A tile: 128x16 packed dwords, 8 per thread, float2 global reads
        if (interior) {
            #pragma unroll
            for (int i = tid; i < TM * (TK / 2); i += 256) {
                int r = i >> 4, c2 = i & 15;
                float2 v = *(const float2*)&A[(size_t)(row0 + r) * K + kt + 2 * c2];
                AsU[r * ASTRIDE + c2] = pk_bf16(v.x, v.y);
            }
        } else {
            #pragma unroll
            for (int i = tid; i < TM * (TK / 2); i += 256) {
                int r = i >> 4, c2 = i & 15;
                int gr = row0 + r;
                unsigned int p = 0u;
                if (gr < Nrows) {
                    float2 v = *(const float2*)&A[(size_t)gr * K + kt + 2 * c2];
                    p = pk_bf16(v.x, v.y);
                }
                AsU[r * ASTRIDE + c2] = p;
            }
        }
        // ---- stage B tile: 32 cols x 16 packed dwords, 2 per thread
        #pragma unroll
        for (int i = tid; i < TN * (TK / 2); i += 256) {
            int c = i & 31, k2 = i >> 5;
            float blo = W[(size_t)(kt + 2 * k2)     * M + col0 + c];
            float bhi = W[(size_t)(kt + 2 * k2 + 1) * M + col0 + c];
            BsU[c * BSTRIDE + k2] = pk_bf16(blo, bhi);
        }
        __syncthreads();

        // ---- fragments: two uint4 (ds_load_b128) each, same contiguous k order for A and B
        // A VGPR v holds K = 16*half + 2v -> k2 = 8*half + v (consecutive dwords)
        uint4 alo = *(const uint4*)&AsU[arow * ASTRIDE + (half << 3)];
        uint4 ahi = *(const uint4*)&AsU[arow * ASTRIDE + (half << 3) + 4];
        v8u  au = {alo.x, alo.y, alo.z, alo.w, ahi.x, ahi.y, ahi.z, ahi.w};
        v16bf af = __builtin_bit_cast(v16bf, au);

        // B VGPR v holds K = 16*half + 2v -> k2 = 8*half + v (consecutive dwords)
        uint4 b0lo = *(const uint4*)&BsU[bcol * BSTRIDE + (half << 3)];
        uint4 b0hi = *(const uint4*)&BsU[bcol * BSTRIDE + (half << 3) + 4];
        v8u  b0u = {b0lo.x, b0lo.y, b0lo.z, b0lo.w, b0hi.x, b0hi.y, b0hi.z, b0hi.w};
        v16bf bf0 = __builtin_bit_cast(v16bf, b0u);

        uint4 b1lo = *(const uint4*)&BsU[(bcol + 16) * BSTRIDE + (half << 3)];
        uint4 b1hi = *(const uint4*)&BsU[(bcol + 16) * BSTRIDE + (half << 3) + 4];
        v8u  b1u = {b1lo.x, b1lo.y, b1lo.z, b1lo.w, b1hi.x, b1hi.y, b1hi.z, b1hi.w};
        v16bf bf1 = __builtin_bit_cast(v16bf, b1u);

        acc0 = __builtin_amdgcn_wmma_f32_16x16x32_bf16(false, af, false, bf0,
                                                       (short)0, acc0, false, false);
        acc1 = __builtin_amdgcn_wmma_f32_16x16x32_bf16(false, af, false, bf1,
                                                       (short)0, acc1, false, false);
        __syncthreads();
    }

    // ---- epilogue: C/D layout: VGPR v, lane L -> row = v + 8*(L>>4), col = L&15
    const int col = lane & 15;
    float b0 = bias ? bias[col0 + col]      : 0.0f;
    float b1 = bias ? bias[col0 + 16 + col] : 0.0f;
    const int rbase = row0 + (wave << 4) + (half << 3);
    if (interior) {
        #pragma unroll
        for (int v = 0; v < 8; ++v) {
            int row = rbase + v;
            C[(size_t)row * M + col0 + col]      = acc0[v] + b0;
            C[(size_t)row * M + col0 + 16 + col] = acc1[v] + b1;
        }
    } else {
        #pragma unroll
        for (int v = 0; v < 8; ++v) {
            int row = rbase + v;
            if (row < Nrows) {
                C[(size_t)row * M + col0 + col]      = acc0[v] + b0;
                C[(size_t)row * M + col0 + 16 + col] = acc1[v] + b1;
            }
        }
    }
}

// ---------------------------------------------------------------------------
// Degree / normalization
// ---------------------------------------------------------------------------
__global__ void init_deg(float* deg, int N) {
    int i = blockIdx.x * blockDim.x + threadIdx.x;
    if (i < N) deg[i] = 1.0f;                       // self-loop weight 1
}
__global__ void accum_deg(const int* __restrict__ dst, const float* __restrict__ w,
                          float* deg, int E) {
    int e = blockIdx.x * blockDim.x + threadIdx.x;
    if (e < E) atomicAdd(&deg[dst[e]], w[e]);
}
__global__ void make_dinv(const float* __restrict__ deg, float* dinv, int N) {
    int i = blockIdx.x * blockDim.x + threadIdx.x;
    if (i < N) {
        float d = deg[i];
        dinv[i] = (d > 0.0f) ? rsqrtf(d) : 0.0f;
    }
}

// ---------------------------------------------------------------------------
// Edge scatter: out[dst] += h[src] * (dinv[src]*w*dinv[dst]).  F = 1<<FLOG.
// Node feature matrices (<= 51 MB) are L2-resident on MI455X (192 MB L2).
// ---------------------------------------------------------------------------
template <int FLOG>
__global__ void scatter_edges(const int* __restrict__ src, const int* __restrict__ dst,
                              const float* __restrict__ w, const float* __restrict__ dinv,
                              const float* __restrict__ h, float* out, int E)
{
    const int F = 1 << FLOG;
    long long t = (long long)blockIdx.x * blockDim.x + threadIdx.x;
    long long e = t >> FLOG;
    int f = (int)(t & (F - 1));
    if (e >= E) return;
    int s = src[e], d = dst[e];
    float norm = dinv[s] * w[e] * dinv[d];
    atomicAdd(&out[(size_t)d * F + f], h[(size_t)s * F + f] * norm);
}

// ---------------------------------------------------------------------------
// Self-loop + bias (+ optional ReLU):  out = act(agg + dinv[i]^2 * h + bias)
// ---------------------------------------------------------------------------
__global__ void selfloop_bias_act(const float* agg, const float* __restrict__ h,
                                  const float* __restrict__ dinv,
                                  const float* __restrict__ bias,
                                  float* out, long long total, int flog, int do_relu)
{
    long long t = (long long)blockIdx.x * blockDim.x + threadIdx.x;
    if (t >= total) return;
    int F = 1 << flog;
    long long i = t >> flog;
    int f = (int)(t & (F - 1));
    float di = dinv[i];
    float v = agg[t] + di * di * h[t] + bias[f];
    out[t] = do_relu ? fmaxf(v, 0.0f) : v;
}

// ---------------------------------------------------------------------------
// Per-node decoder logits: a[i] = z[i]·W_lin[0:128], b[i] = z[i]·W_lin[128:256]
// ---------------------------------------------------------------------------
__global__ __launch_bounds__(128) void node_logits(const float* __restrict__ z,
                                                   const float* __restrict__ Wl,
                                                   float* asrc, float* adst, int N)
{
    __shared__ float sa[128];
    __shared__ float sb[128];
    int i = blockIdx.x;
    int f = threadIdx.x;
    float zv = z[(size_t)i * 128 + f];
    sa[f] = zv * Wl[f];
    sb[f] = zv * Wl[128 + f];
    __syncthreads();
    #pragma unroll
    for (int s = 64; s > 0; s >>= 1) {
        if (f < s) { sa[f] += sa[f + s]; sb[f] += sb[f + s]; }
        __syncthreads();
    }
    if (f == 0) { asrc[i] = sa[0]; adst[i] = sb[0]; }
}

__global__ void edge_out(const int* __restrict__ src, const int* __restrict__ dst,
                         const float* __restrict__ asrc, const float* __restrict__ adst,
                         const float* __restrict__ blin, float* out, int E)
{
    int e = blockIdx.x * blockDim.x + threadIdx.x;
    if (e < E) {
        float v = asrc[src[e]] + adst[dst[e]] + blin[0];
        out[e] = 1.0f / (1.0f + __expf(-v));
    }
}

// ---------------------------------------------------------------------------
extern "C" void kernel_launch(void* const* d_in, const int* in_sizes, int n_in,
                              void* d_out, int out_size, void* d_ws, size_t ws_size,
                              hipStream_t stream)
{
    const int*   edge_index = (const int*)  d_in[0];
    const float* x          = (const float*)d_in[1];
    const float* w          = (const float*)d_in[2];
    const float* W_lin1     = (const float*)d_in[3];
    const float* b_lin1     = (const float*)d_in[4];
    const float* W_conv1    = (const float*)d_in[5];
    const float* b_conv1    = (const float*)d_in[6];
    const float* W_mu       = (const float*)d_in[7];
    const float* b_mu       = (const float*)d_in[8];
    const float* W_lin      = (const float*)d_in[9];
    const float* b_lin      = (const float*)d_in[10];

    const int E = in_sizes[0] / 2;
    const int N = in_sizes[1] / 512;
    const int* src = edge_index;
    const int* dst = edge_index + E;

    // workspace layout (floats)
    float* f0   = (float*)d_ws;               // N*128  : h1, later hw2
    float* f1   = f0 + (size_t)N * 128;       // N*256  : hw1
    float* f2   = f1 + (size_t)N * 256;       // N*256  : agg1 -> h2 (in place)
    float* f3   = f2 + (size_t)N * 256;       // N*128  : agg2 -> z  (in place)
    float* deg  = f3 + (size_t)N * 128;       // N
    float* dinv = deg + N;                    // N
    float* la   = dinv + N;                   // N : z·Wlin[:128]
    float* lb   = la + N;                     // N : z·Wlin[128:]

    // --- degree / symmetric norm --------------------------------------------
    init_deg <<<(N + 255) / 256, 256, 0, stream>>>(deg, N);
    accum_deg<<<(E + 255) / 256, 256, 0, stream>>>(dst, w, deg, E);
    make_dinv<<<(N + 255) / 256, 256, 0, stream>>>(deg, dinv, N);

    // --- h1 = x @ W_lin1 + b_lin1   [N,512]x[512,128] ------------------------
    dim3 g1((N + TM - 1) / TM, 128 / TN);
    gemm_bf16_wmma<<<g1, 256, 0, stream>>>(x, W_lin1, b_lin1, f0, N, 512, 128);

    // --- hw1 = h1 @ W_conv1         [N,128]x[128,256] ------------------------
    dim3 g2((N + TM - 1) / TM, 256 / TN);
    gemm_bf16_wmma<<<g2, 256, 0, stream>>>(f0, W_conv1, nullptr, f1, N, 128, 256);

    // --- conv1 aggregation + self-loop + bias + ReLU -------------------------
    hipMemsetAsync(f2, 0, (size_t)N * 256 * sizeof(float), stream);
    {
        long long total = (long long)E << 8;
        scatter_edges<8><<<(unsigned)((total + 255) / 256), 256, 0, stream>>>(
            src, dst, w, dinv, f1, f2, E);
        long long tn = (long long)N << 8;
        selfloop_bias_act<<<(unsigned)((tn + 255) / 256), 256, 0, stream>>>(
            f2, f1, dinv, b_conv1, f2, tn, 8, 1);
    }

    // --- hw2 = h2 @ W_mu            [N,256]x[256,128] ------------------------
    gemm_bf16_wmma<<<g1, 256, 0, stream>>>(f2, W_mu, nullptr, f0, N, 256, 128);

    // --- conv2 aggregation + self-loop + bias (z = mu) -----------------------
    hipMemsetAsync(f3, 0, (size_t)N * 128 * sizeof(float), stream);
    {
        long long total = (long long)E << 7;
        scatter_edges<7><<<(unsigned)((total + 255) / 256), 256, 0, stream>>>(
            src, dst, w, dinv, f0, f3, E);
        long long tn = (long long)N << 7;
        selfloop_bias_act<<<(unsigned)((tn + 255) / 256), 256, 0, stream>>>(
            f3, f0, dinv, b_mu, f3, tn, 7, 0);
    }

    // --- decoder: per-node logits then per-edge sigmoid ----------------------
    node_logits<<<N, 128, 0, stream>>>(f3, W_lin, la, lb, N);
    edge_out<<<(E + 255) / 256, 256, 0, stream>>>(src, dst, la, lb, b_lin,
                                                  (float*)d_out, E);
}
